// VQEmbedding_36618891166243
// MI455X (gfx1250) — matrix operand order repaired and checked by reading, not compiled, not run on hardware
//
#include <hip/hip_runtime.h>

typedef __attribute__((ext_vector_type(16))) __bf16 v16bf;
typedef __attribute__((ext_vector_type(8)))  float  v8f;

#define K_EMB 1024
#define D_EMB 256
#define ROWS_PER_BLOCK 128   // 8 waves x 16 rows
#define LDS_STRIDE 272       // 256 + 16 pad (ushort units) to dodge bank conflicts

// ---------------- prep: codebook fp32 -> bf16, and e_sq[k] ----------------
__global__ __launch_bounds__(256) void vq_prep(const float* __restrict__ cb,
                                               unsigned short* __restrict__ cb_bf,
                                               float* __restrict__ esq) {
    int code = blockIdx.x;          // 0..1023
    int t    = threadIdx.x;         // 0..255 == d
    float v = cb[(size_t)code * D_EMB + t];
    __bf16 h = (__bf16)v;
    cb_bf[(size_t)code * D_EMB + t] = __builtin_bit_cast(unsigned short, h);
    float s = v * v;
#pragma unroll
    for (int m = 16; m >= 1; m >>= 1) s += __shfl_xor(s, m, 32);
    __shared__ float part[8];
    int lane = t & 31, w = t >> 5;
    if (lane == 0) part[w] = s;
    __syncthreads();
    if (t == 0) {
        float tot = 0.f;
#pragma unroll
        for (int i = 0; i < 8; ++i) tot += part[i];
        esq[code] = tot;
    }
}

// ---------------- main: WMMA distances + argmin + gather ----------------
__global__ __launch_bounds__(256) void vq_main(const float* __restrict__ z,
                                               const float* __restrict__ codebook,
                                               const unsigned short* __restrict__ cb_bf,
                                               const float* __restrict__ esq,
                                               float* __restrict__ zq_out,
                                               float* __restrict__ idx_out) {
    __shared__ __attribute__((aligned(32))) unsigned short lds[16 * LDS_STRIDE];

    const int tid  = threadIdx.x;
    const int lane = tid & 31;
    const int wave = tid >> 5;
    const int col  = lane & 15;   // WMMA B/C column (N), also z row within wave tile
    const int hi   = lane >> 4;   // half-wave select (K-split / M-split)
    const int row_base = blockIdx.x * ROWS_PER_BLOCK + wave * 16;
    const int my_row   = row_base + col;

    // ---- A fragments: 16 rows x 256 K of bf16, resident in registers ----
    // A 16x32 bf16, contiguous k order used by this code (not ISA table order): lane<16 elems0..15 = K 0..15;
    //                      lane>=16 elems0..15 = K16..31.
    v16bf a[8];
    {
        const float* zr = z + (size_t)my_row * D_EMB;
#pragma unroll
        for (int t = 0; t < 8; ++t) {
            const int d0 = t * 32 + hi * 16;
            float4 f0 = *(const float4*)(zr + d0);
            float4 f1 = *(const float4*)(zr + d0 + 4);
            float4 f2 = *(const float4*)(zr + d0 + 8);
            float4 f3 = *(const float4*)(zr + d0 + 12);
            v16bf av;
            av[0]=(__bf16)f0.x;  av[1]=(__bf16)f0.y;  av[2]=(__bf16)f0.z;  av[3]=(__bf16)f0.w;
            av[4]=(__bf16)f1.x;  av[5]=(__bf16)f1.y;  av[6]=(__bf16)f1.z;  av[7]=(__bf16)f1.w;
            av[8]=(__bf16)f2.x;  av[9]=(__bf16)f2.y;  av[10]=(__bf16)f2.z; av[11]=(__bf16)f2.w;
            av[12]=(__bf16)f3.x; av[13]=(__bf16)f3.y; av[14]=(__bf16)f3.z; av[15]=(__bf16)f3.w;
            a[t] = av;
        }
    }

    float minv[8];
    int   mini[8];
#pragma unroll
    for (int v = 0; v < 8; ++v) { minv[v] = 3.402823466e38f; mini[v] = 0; }

    // staging assignment: 256 threads cover 16 codes x 16 chunks of 16 bf16
    const int s_code  = tid >> 4;   // 0..15
    const int s_chunk = tid & 15;   // 0..15

    for (int kb = 0; kb < 64; ++kb) {
        __syncthreads();            // previous tile fully consumed
        {   // stage 16 codes x 256 d (bf16, 8KB) into padded LDS
            const int4* src = (const int4*)(cb_bf +
                ((size_t)(kb * 16 + s_code) * D_EMB + s_chunk * 16));
            int4* dst = (int4*)(lds + s_code * LDS_STRIDE + s_chunk * 16);
            dst[0] = src[0];
            dst[1] = src[1];
        }
        __syncthreads();

        const float e2    = esq[kb * 16 + col];
        const int   code0 = kb * 16 + col;

        v8f acc = {};
#pragma unroll
        for (int t = 0; t < 8; ++t) {
            // B 32x16 bf16, contiguous k order used by this code (matches A): lane = column, K packed along VGPRs;
            // lanes>=16 carry K 16..31 of the same column.
            v16bf b = *(const v16bf*)(lds + col * LDS_STRIDE + t * 32 + hi * 16);
            acc = __builtin_amdgcn_wmma_f32_16x16x32_bf16(
                false, a[t], false, b, (short)0, acc, false, false);
        }

        // dist proxy: e_sq[k] - 2*cross  (||z||^2 constant per row, dropped)
#pragma unroll
        for (int v = 0; v < 8; ++v) {
            float dd = fmaf(-2.f, acc[v], e2);
            if (dd < minv[v]) { minv[v] = dd; mini[v] = code0; }
        }
    }

    // ---- argmin reduce across the 16 columns (per half-wave) ----
#pragma unroll
    for (int v = 0; v < 8; ++v) {
#pragma unroll
        for (int m = 1; m <= 8; m <<= 1) {
            float ov = __shfl_xor(minv[v], m, 32);
            int   oi = __shfl_xor(mini[v], m, 32);
            if (ov < minv[v] || (ov == minv[v] && oi < mini[v])) {
                minv[v] = ov; mini[v] = oi;
            }
        }
    }
    // now lanes 0..15 all hold winners for rows 0..7 (v), lanes 16..31 for rows 8..15

    if (col == 0) {   // lanes 0 and 16 write 8 consecutive indices each
#pragma unroll
        for (int v = 0; v < 8; ++v)
            idx_out[row_base + hi * 8 + v] = (float)mini[v];
    }

    // ---- gather winning codebook rows (fp32) to z_q, whole wave per row ----
#pragma unroll
    for (int m = 0; m < 16; ++m) {
        const int src_lane = (m < 8) ? 0 : 16;
        const int idx = __shfl(mini[m & 7], src_lane, 32);
        const float4* s = (const float4*)(codebook + (size_t)idx * D_EMB) + lane * 2;
        float4*       d = (float4*)(zq_out + (size_t)(row_base + m) * D_EMB) + lane * 2;
        d[0] = s[0];
        d[1] = s[1];
    }
}

extern "C" void kernel_launch(void* const* d_in, const int* in_sizes, int n_in,
                              void* d_out, int out_size, void* d_ws, size_t ws_size,
                              hipStream_t stream) {
    const float* z  = (const float*)d_in[0];   // [32,4096,256] fp32
    const float* cb = (const float*)d_in[1];   // [1024,256] fp32

    unsigned short* cb_bf = (unsigned short*)d_ws;                       // 512 KB
    float* esq = (float*)((char*)d_ws + (size_t)K_EMB * D_EMB * 2);      // 4 KB

    float* zq  = (float*)d_out;                                          // [M,256]
    float* idx = zq + (size_t)32 * 4096 * D_EMB;                         // [M] as float

    vq_prep<<<K_EMB, 256, 0, stream>>>(cb, cb_bf, esq);

    const int rows = in_sizes[0] / D_EMB;      // 131072
    vq_main<<<rows / ROWS_PER_BLOCK, 256, 0, stream>>>(z, cb, cb_bf, esq, zq, idx);
}
